// InteractionBlock_59450937311948
// MI455X (gfx1250) — matrix operand order repaired and checked by reading, not compiled, not run on hardware
//
#include <hip/hip_runtime.h>
#include <cstdint>
#include <cstddef>

// ---------------------------------------------------------------------------
// SchNet interaction block for MI455X (gfx1250), bf16 WMMA + async LDS copies.
//
//   h   = x @ fc1_w.T                                  (node GEMM, 50k rows)
//   W   = ssp(ssp(ea @ f_w1.T + b1) @ f_w2.T + b2)     (edge MLP, fused)
//   agg = scatter_add(h[src] * W, dst)                 (atomic f32 into L2)
//   out = ssp(agg @ s_w1.T) @ s_w2.T                   (node MLP, fused)
//
// All GEMMs: M-tile = 16 rows per wave, N = 128 (8 WMMA col-tiles), K = 128
// (4 slabs of 32).  v_wmma_f32_16x16x32_bf16, f32 accumulation.
// ---------------------------------------------------------------------------

typedef __attribute__((ext_vector_type(16))) __bf16 v16bf;
typedef __attribute__((ext_vector_type(8)))  float  v8f;

#define WPB 4          // waves per block
#define LOG2F_ 0.6931471805599453f

__device__ __forceinline__ float ssp_f(float x) {
    // softplus(x) - log(2), numerically stable
    float ax = __builtin_fabsf(x);
    float e  = __builtin_expf(-ax);
    return __builtin_fmaxf(x, 0.0f) + __builtin_logf(1.0f + e) - LOG2F_;
}

// --- CDNA5 async global->LDS copy (16B per lane per issue), ASYNCcnt-tracked
__device__ __forceinline__ void async_copy_b128(uint32_t lds_off, const void* gaddr) {
    asm volatile("global_load_async_to_lds_b128 %0, %1, off"
                 :: "v"(lds_off), "v"(gaddr) : "memory");
}
__device__ __forceinline__ void wait_async() {
    asm volatile("s_wait_asynccnt 0" ::: "memory");
}
// --- hardware f32 atomic add (no return) into global memory
__device__ __forceinline__ void atomic_add_f32g(float* p, float v) {
    asm volatile("global_atomic_add_f32 %0, %1, off" :: "v"(p), "v"(v) : "memory");
}
// generic pointer to LDS object -> wave-relative LDS byte offset (addr[31:0])
__device__ __forceinline__ uint32_t lds_off_of(const void* p) {
    return (uint32_t)(uintptr_t)p;
}

// B fragment: 32(K) x 16(N) bf16 tile of a weight matrix stored row-major
// [n=0..127][k=0..127] in LDS.  Lanes 0-15: col n, K=slab+0..15;
// lanes 16-31: col n, K=slab+16..31 (contiguous k order, same as the A frags).
__device__ __forceinline__ v16bf b_frag(const __bf16* w, int lane, int nt, int kb) {
    int n     = nt * 16 + (lane & 15);
    int kbase = kb * 32 + ((lane >> 4) << 4);
    return *(const v16bf*)(w + n * 128 + kbase);
}

// A fragment from an f32 row-major tile (stride floats per row), converting to
// bf16.  Lane layout: lanes 0-15 row m, elems 0..15 -> K+0..15;
// lanes 16-31 row m, elems 0..15 -> K+16..31 (matches b_frag).
__device__ __forceinline__ v16bf a_frag_f32(const float* a, int stride, int lane, int kb) {
    int m  = lane & 15;
    int c0 = kb * 32 + ((lane >> 4) << 4);
    const float* r0 = a + m * stride + c0;
    v16bf f;
#pragma unroll
    for (int j = 0; j < 8; ++j) {
        f[j]     = (__bf16)r0[j];
        f[j + 8] = (__bf16)r0[8 + j];
    }
    return f;
}

// Same but A already bf16 in LDS (padded stride to dodge bank conflicts).
__device__ __forceinline__ v16bf a_frag_bf16(const __bf16* a, int stride, int lane, int kb) {
    int m  = lane & 15;
    int c0 = kb * 32 + ((lane >> 4) << 4);
    const __bf16* r0 = a + m * stride + c0;
    v16bf f;
#pragma unroll
    for (int j = 0; j < 8; ++j) {
        f[j]     = r0[j];
        f[j + 8] = r0[8 + j];
    }
    return f;
}

__device__ __forceinline__ void zero_acc(v8f acc[8]) {
#pragma unroll
    for (int i = 0; i < 8; ++i)
        acc[i] = (v8f){0.f, 0.f, 0.f, 0.f, 0.f, 0.f, 0.f, 0.f};
}

// 16x128 += (16x128 f32 tile) @ W.T  with W bf16 in LDS
__device__ __forceinline__ void gemm_tile_f32(const float* aStage, int stride,
                                              const __bf16* w, int lane, v8f acc[8]) {
#pragma unroll
    for (int kb = 0; kb < 4; ++kb) {
        v16bf a = a_frag_f32(aStage, stride, lane, kb);
#pragma unroll
        for (int nt = 0; nt < 8; ++nt) {
            v16bf b = b_frag(w, lane, nt, kb);
            acc[nt] = __builtin_amdgcn_wmma_f32_16x16x32_bf16(
                false, a, false, b, (short)0, acc[nt], false, false);
        }
    }
}

__device__ __forceinline__ void gemm_tile_bf16(const __bf16* aStage, int stride,
                                               const __bf16* w, int lane, v8f acc[8]) {
#pragma unroll
    for (int kb = 0; kb < 4; ++kb) {
        v16bf a = a_frag_bf16(aStage, stride, lane, kb);
#pragma unroll
        for (int nt = 0; nt < 8; ++nt) {
            v16bf b = b_frag(w, lane, nt, kb);
            acc[nt] = __builtin_amdgcn_wmma_f32_16x16x32_bf16(
                false, a, false, b, (short)0, acc[nt], false, false);
        }
    }
}

// ---------------------------------------------------------------------------
// Kernel 1: h = x @ fc1_w.T          dyn-LDS: 32768 (w) + WPB*8192 (stage)
// ---------------------------------------------------------------------------
__global__ void k_node_fc1(const float* __restrict__ x, const float* __restrict__ wglob,
                           float* __restrict__ h, int ntiles) {
    extern __shared__ v16bf dynsmem[];
    char* smem = (char*)dynsmem;
    __bf16* wlds = (__bf16*)smem;
    const int tid = threadIdx.x, wv = tid >> 5, lane = tid & 31;
    float* stA = (float*)(smem + 32768 + wv * 8192);

    for (int i = tid; i < 128 * 128; i += blockDim.x) wlds[i] = (__bf16)wglob[i];
    __syncthreads();

    int tile = blockIdx.x * WPB + wv;
    if (tile >= ntiles) return;
    int m0 = tile * 16;

    const char* g   = (const char*)(x + (size_t)m0 * 128);
    uint32_t    la  = lds_off_of(stA);
#pragma unroll
    for (int it = 0; it < 16; ++it)
        async_copy_b128(la + (uint32_t)(it * 32 + lane) * 16u,
                        g + (size_t)(it * 32 + lane) * 16);
    wait_async();

    v8f acc[8];
    zero_acc(acc);
    gemm_tile_f32(stA, 128, wlds, lane, acc);

    int mo = (lane >> 4) * 8, n0 = lane & 15;
#pragma unroll
    for (int nt = 0; nt < 8; ++nt)
#pragma unroll
        for (int r = 0; r < 8; ++r)
            h[(size_t)(m0 + mo + r) * 128 + nt * 16 + n0] = acc[nt][r];
}

// ---------------------------------------------------------------------------
// Kernel 2 (fused edge pipeline):
//   W = ssp(ssp(ea@w1.T+b1)@w2.T+b2);  atomic agg[dst] += h[src] * W
// dyn-LDS: 65536 (w1,w2) + WPB*20992 (stA 8K | stH 8K | stT 4.25K+pad)
// ---------------------------------------------------------------------------
__global__ void k_edge(const float* __restrict__ ea, const int* __restrict__ idx,
                       const float* __restrict__ h,
                       const float* __restrict__ w1g, const float* __restrict__ b1,
                       const float* __restrict__ w2g, const float* __restrict__ b2,
                       float* __restrict__ agg, int E, int etiles) {
    extern __shared__ v16bf dynsmem[];
    char* smem = (char*)dynsmem;
    __bf16* w1 = (__bf16*)smem;
    __bf16* w2 = (__bf16*)(smem + 32768);
    const int tid = threadIdx.x, wv = tid >> 5, lane = tid & 31;
    char*   wsb = smem + 65536 + wv * 20992;
    float*  stA = (float*)wsb;                 // 16x128 f32 edge_attr tile
    float*  stH = (float*)(wsb + 8192);        // 16x128 f32 gathered h rows
    __bf16* stT = (__bf16*)(wsb + 16384);      // 16x136 bf16 t1 (padded)

    for (int i = tid; i < 128 * 128; i += blockDim.x) {
        w1[i] = (__bf16)w1g[i];
        w2[i] = (__bf16)w2g[i];
    }
    __syncthreads();

    int tile = blockIdx.x * WPB + wv;
    if (tile >= etiles) return;
    int e0 = tile * 16;
    const int* srcI = idx;
    const int* dstI = idx + E;

    // async-stage the contiguous 8KB edge_attr tile
    const char* g  = (const char*)(ea + (size_t)e0 * 128);
    uint32_t    la = lds_off_of(stA), lh = lds_off_of(stH);
#pragma unroll
    for (int it = 0; it < 16; ++it)
        async_copy_b128(la + (uint32_t)(it * 32 + lane) * 16u,
                        g + (size_t)(it * 32 + lane) * 16);
    // async-gather the 16 source-node rows of h (each row = 512B = 32 lanes x 16B)
#pragma unroll
    for (int i = 0; i < 16; ++i) {
        int sr = srcI[e0 + i];
        async_copy_b128(lh + (uint32_t)(i * 512 + lane * 16),
                        (const char*)(h + (size_t)sr * 128) + lane * 16);
    }
    wait_async();

    // GEMM1 + bias + ssp  ->  stT (bf16, transposed via LDS for GEMM2 A-layout)
    v8f acc[8];
    zero_acc(acc);
    gemm_tile_f32(stA, 128, w1, lane, acc);
    int mo = (lane >> 4) * 8, n0 = lane & 15;
#pragma unroll
    for (int nt = 0; nt < 8; ++nt) {
        float bias = b1[nt * 16 + n0];
#pragma unroll
        for (int r = 0; r < 8; ++r) {
            float t = ssp_f(acc[nt][r] + bias);
            stT[(mo + r) * 136 + nt * 16 + n0] = (__bf16)t;
        }
    }
    // same-wave LDS ops are ordered (DScnt) -> no barrier needed

    // GEMM2 + bias + ssp = filter W; multiply by gathered h; scatter-add
    v8f acc2[8];
    zero_acc(acc2);
    gemm_tile_bf16(stT, 136, w2, lane, acc2);

    int drow[8];
#pragma unroll
    for (int r = 0; r < 8; ++r) drow[r] = dstI[e0 + mo + r];

#pragma unroll
    for (int nt = 0; nt < 8; ++nt) {
        float bias = b2[nt * 16 + n0];
#pragma unroll
        for (int r = 0; r < 8; ++r) {
            float wf  = ssp_f(acc2[nt][r] + bias);
            float msg = wf * stH[(mo + r) * 128 + nt * 16 + n0];
            atomic_add_f32g(agg + (size_t)drow[r] * 128 + nt * 16 + n0, msg);
        }
    }
}

// ---------------------------------------------------------------------------
// Kernel 3: out = ssp(agg @ s_w1.T) @ s_w2.T
// dyn-LDS: 65536 (w1,w2) + WPB*12544 (stA 8K | stT 4.25K+pad)
// ---------------------------------------------------------------------------
__global__ void k_node_out(const float* __restrict__ agg,
                           const float* __restrict__ w1g, const float* __restrict__ w2g,
                           float* __restrict__ out, int ntiles) {
    extern __shared__ v16bf dynsmem[];
    char* smem = (char*)dynsmem;
    __bf16* w1 = (__bf16*)smem;
    __bf16* w2 = (__bf16*)(smem + 32768);
    const int tid = threadIdx.x, wv = tid >> 5, lane = tid & 31;
    char*   wsb = smem + 65536 + wv * 12544;
    float*  stA = (float*)wsb;
    __bf16* stT = (__bf16*)(wsb + 8192);

    for (int i = tid; i < 128 * 128; i += blockDim.x) {
        w1[i] = (__bf16)w1g[i];
        w2[i] = (__bf16)w2g[i];
    }
    __syncthreads();

    int tile = blockIdx.x * WPB + wv;
    if (tile >= ntiles) return;
    int m0 = tile * 16;

    const char* g  = (const char*)(agg + (size_t)m0 * 128);
    uint32_t    la = lds_off_of(stA);
#pragma unroll
    for (int it = 0; it < 16; ++it)
        async_copy_b128(la + (uint32_t)(it * 32 + lane) * 16u,
                        g + (size_t)(it * 32 + lane) * 16);
    wait_async();

    v8f acc[8];
    zero_acc(acc);
    gemm_tile_f32(stA, 128, w1, lane, acc);
    int mo = (lane >> 4) * 8, n0 = lane & 15;
#pragma unroll
    for (int nt = 0; nt < 8; ++nt)
#pragma unroll
        for (int r = 0; r < 8; ++r)
            stT[(mo + r) * 136 + nt * 16 + n0] = (__bf16)ssp_f(acc[nt][r]);

    v8f acc2[8];
    zero_acc(acc2);
    gemm_tile_bf16(stT, 136, w2, lane, acc2);
#pragma unroll
    for (int nt = 0; nt < 8; ++nt)
#pragma unroll
        for (int r = 0; r < 8; ++r)
            out[(size_t)(m0 + mo + r) * 128 + nt * 16 + n0] = acc2[nt][r];
}

// ---------------------------------------------------------------------------
extern "C" void kernel_launch(void* const* d_in, const int* in_sizes, int n_in,
                              void* d_out, int out_size, void* d_ws, size_t ws_size,
                              hipStream_t stream) {
    const float* x     = (const float*)d_in[0];
    const float* ea    = (const float*)d_in[1];
    const int*   ei    = (const int*)d_in[2];
    const float* fc1_w = (const float*)d_in[3];
    const float* f_w1  = (const float*)d_in[4];
    const float* f_b1  = (const float*)d_in[5];
    const float* f_w2  = (const float*)d_in[6];
    const float* f_b2  = (const float*)d_in[7];
    const float* s_w1  = (const float*)d_in[8];
    const float* s_w2  = (const float*)d_in[9];
    float*       out   = (float*)d_out;

    const int N = in_sizes[0] / 128;   // 50000
    const int E = in_sizes[1] / 128;   // 600000

    // workspace: h [N,128] f32 then agg [N,128] f32  (~51.2 MB)
    float* h   = (float*)d_ws;
    float* agg = h + (size_t)N * 128;

    hipMemsetAsync(agg, 0, (size_t)N * 128 * sizeof(float), stream);

    const int ntiles = (N + 15) / 16;
    const int etiles = (E + 15) / 16;
    dim3 blk(32 * WPB);

    const size_t lds1 = 32768 + WPB * 8192;            // 64 KB
    const size_t lds2 = 65536 + WPB * 20992;           // ~146 KB (of 320 KB WGP)
    const size_t lds3 = 65536 + WPB * 12544;           // ~113 KB

    k_node_fc1<<<(ntiles + WPB - 1) / WPB, blk, lds1, stream>>>(x, fc1_w, h, ntiles);
    k_edge<<<(etiles + WPB - 1) / WPB, blk, lds2, stream>>>(
        ea, ei, h, f_w1, f_b1, f_w2, f_b2, agg, E, etiles);
    k_node_out<<<(ntiles + WPB - 1) / WPB, blk, lds3, stream>>>(agg, s_w1, s_w2, out, ntiles);
}
